// PillarMotionNet_59141699666576
// MI455X (gfx1250) — matrix operand order repaired and checked by reading, not compiled, not run on hardware
//
#include <hip/hip_runtime.h>
#include <stdint.h>

// ---------------- CDNA5 feature detection ----------------
#if __has_builtin(__builtin_amdgcn_global_load_async_to_lds_b128)
#define HAVE_ASYNC_LOAD 1
#endif
#if __has_builtin(__builtin_amdgcn_global_store_async_from_lds_b128)
#define HAVE_ASYNC_STORE 1
#endif
#if __has_builtin(__builtin_amdgcn_s_wait_asynccnt)
#define HAVE_WAIT_ASYNC 1
#endif

typedef __attribute__((ext_vector_type(16))) _Float16 v16h;
typedef __attribute__((ext_vector_type(8)))  float    v8f;
typedef __attribute__((ext_vector_type(4)))  int      v4i;

#define GX 400
#define GY 400
#define NT 5
#define NKEYS 3200000              // 4 * 400 * 400 * 5
#define SCAN_TILE 2048             // bins per scan block (8 waves * 256)
#define NBLK2 ((NKEYS + SCAN_TILE - 1) / SCAN_TILE)   // 1563
#define TILE_PTS 512               // points per staged tile (12288 B in, 18432 B out)
#define TPB 4                      // tiles per block in k_accum (double-buffered)

// ---------------- helpers ----------------
__device__ __forceinline__ void atomic_add_f32(float* p, float v) {
  __hip_atomic_fetch_add(p, v, __ATOMIC_RELAXED, __HIP_MEMORY_SCOPE_AGENT);
}
__device__ __forceinline__ void atomic_add_u32(uint32_t* p, uint32_t v) {
  __hip_atomic_fetch_add(p, v, __ATOMIC_RELAXED, __HIP_MEMORY_SCOPE_AGENT);
}

// 16B global -> LDS, async (ASYNCcnt tracked); sync fallback otherwise.
__device__ __forceinline__ void stage16(const void* g, void* lds) {
#if defined(HAVE_ASYNC_LOAD)
  __builtin_amdgcn_global_load_async_to_lds_b128(
      (__attribute__((address_space(1))) v4i*)(uintptr_t)g,
      (__attribute__((address_space(3))) v4i*)lds, 0, 0);
#else
  *(float4*)lds = *(const float4*)g;
#endif
}
// 16B LDS -> global, async; sync fallback otherwise.
__device__ __forceinline__ void drain16(void* lds, void* g) {
#if defined(HAVE_ASYNC_STORE)
  __builtin_amdgcn_global_store_async_from_lds_b128(
      (__attribute__((address_space(1))) v4i*)(uintptr_t)g,
      (__attribute__((address_space(3))) v4i*)lds, 0, 0);
#else
  *(float4*)g = *(const float4*)lds;
#endif
}
__device__ __forceinline__ void wait_async0() {
#if defined(HAVE_WAIT_ASYNC)
  __builtin_amdgcn_s_wait_asynccnt(0);
#elif defined(HAVE_ASYNC_LOAD)
  asm volatile("s_wait_asynccnt 0" ::: "memory");
#endif
}
__device__ __forceinline__ void wait_async3() {   // allow 3 outstanding (prefetch)
#if defined(HAVE_WAIT_ASYNC)
  __builtin_amdgcn_s_wait_asynccnt(3);
#elif defined(HAVE_ASYNC_LOAD)
  asm volatile("s_wait_asynccnt 3" ::: "memory");
#endif
}

__device__ __forceinline__ int point_key(const float* p, int& cx, int& cy) {
  int b = (int)p[0];
  cx = (int)((p[1] + 50.0f) * 4.0f);   // 1/0.25 exact
  cy = (int)((p[2] + 50.0f) * 4.0f);
  int t = (int)p[5];
  return ((b * GX + cx) * GY + cy) * NT + t;
}

// ---------------- K0: zero cnt + 3 sums (4*NKEYS words) ----------------
__global__ void k_zero(uint32_t* __restrict__ buf, int nwords) {
  int i = blockIdx.x * blockDim.x + threadIdx.x;
  int stride = gridDim.x * blockDim.x;
  for (; i < nwords; i += stride) buf[i] = 0u;
}

// ---------------- K1: histogram + xyz scatter-add --------------------------
// Double-buffered async global->LDS pipeline: while processing tile t, tile
// t+1 streams into the other LDS buffer; s_wait_asynccnt<=3 releases tile t
// (async loads complete in order).
__global__ __launch_bounds__(256) void k_accum(const float* __restrict__ pts, int n,
                                               uint32_t* __restrict__ cnt,
                                               float* __restrict__ sx,
                                               float* __restrict__ sy,
                                               float* __restrict__ sz) {
  __shared__ float buf[2][TILE_PTS * 6];     // 2 x 12288 B
  int tid = threadIdx.x;
  int tilesTotal = (n + TILE_PTS - 1) / TILE_PTS;
  int t0 = blockIdx.x * TPB;
  if (t0 >= tilesTotal) return;

  // issue prefetch for first tile if it is full
  bool curPref = ((t0 + 1) * TILE_PTS <= n);
  if (curPref) {
    const char* g = (const char*)pts + (size_t)t0 * TILE_PTS * 24;
    char* l = (char*)buf[0];
#pragma unroll
    for (int k = 0; k < 3; ++k) {
      int off = tid * 16 + k * 4096;
      stage16(g + off, l + off);
    }
  }

  for (int it = 0; it < TPB; ++it) {
    int tile = t0 + it;
    if (tile >= tilesTotal) break;
    int base = tile * TILE_PTS;
    int npts = n - base; if (npts > TILE_PTS) npts = TILE_PTS;
    int bi = it & 1;

    // prefetch next tile into the other buffer (safe: it was fully consumed
    // before the barrier that ended the previous iteration)
    int nxt = tile + 1;
    bool nxtPref = (it + 1 < TPB) && (nxt < tilesTotal) && ((nxt + 1) * TILE_PTS <= n);
    if (nxtPref) {
      const char* g = (const char*)pts + (size_t)nxt * TILE_PTS * 24;
      char* l = (char*)buf[(it + 1) & 1];
#pragma unroll
      for (int k = 0; k < 3; ++k) {
        int off = tid * 16 + k * 4096;
        stage16(g + off, l + off);
      }
    }

    if (curPref) {
      if (nxtPref) wait_async3(); else wait_async0();
      __syncthreads();
    } else {
      // tail tile: plain coalesced loads
      for (int e = tid; e < npts * 6; e += 256) buf[bi][e] = pts[(size_t)base * 6 + e];
      __syncthreads();
    }

    for (int p = tid; p < npts; p += 256) {
      const float* q = &buf[bi][p * 6];
      int cx, cy;
      int key = point_key(q, cx, cy);
      atomic_add_u32(&cnt[key], 1u);
      atomic_add_f32(&sx[key], q[1]);
      atomic_add_f32(&sy[key], q[2]);
      atomic_add_f32(&sz[key], q[3]);
    }
    __syncthreads();          // all reads of buf[bi] done before it is refilled
    curPref = nxtPref;
  }
}

// ---------------- K2a: per-block occupancy scan via WMMA -----------------
// Each wave scans 256 bins: D = A x U (16x16x32 f16 WMMA), A = occupancy
// 16x16 (cols 16..31 zero), U = upper-triangular ones -> inclusive row
// prefixes; rowsums combined with a wave32 shuffle scan; waves combined in
// LDS. Exact: all values < 2^24, f16 operands are {0,1}.
__global__ __launch_bounds__(256) void k_scan1(const uint32_t* __restrict__ cnt,
                                               int* __restrict__ rank,
                                               float* __restrict__ partials) {
  __shared__ float occ[SCAN_TILE];
  __shared__ float rowsum[8 * 16];
  __shared__ float rowoff[8 * 16];
  __shared__ float waveoff[8];

  int tid = threadIdx.x;
  int blkBase = blockIdx.x * SCAN_TILE;
#pragma unroll
  for (int k = 0; k < 8; ++k) {
    int li = tid + k * 256;
    int g = blkBase + li;
    occ[li] = (g < NKEYS && cnt[g] != 0u) ? 1.0f : 0.0f;
  }
  __syncthreads();

  int w = tid >> 5;
  int lane = tid & 31;
  int laneN = lane & 15;
  int hi = lane >> 4;
  int wbase = w * 256;

  // A-matrix (16x32 f16 layout): lane holds row M=laneN, K-halfspace by hi.
  v16h a;
  int Kb = hi ? 8 : 0;
#pragma unroll
  for (int h = 0; h < 16; ++h) {
    int j = h >> 1, hf = h & 1;
    int K = 16 * hi + h;
    a[h] = (K < 16) ? (_Float16)occ[wbase + laneN * 16 + K] : (_Float16)0.0f;
  }
  // B-matrix (32x16 f16): lanes 0-15 hold K=0..15 for N=lane; upper-tri ones.
  v16h b;
#pragma unroll
  for (int h = 0; h < 16; ++h)
    b[h] = (hi == 0 && h <= laneN) ? (_Float16)1.0f : (_Float16)0.0f;

  v8f c = {};
  v8f d = __builtin_amdgcn_wmma_f32_16x16x32_f16(false, a, false, b,
                                                 (short)0, c, false, false);

  // D(i,j): lane holds rows i = v + 8*hi, col j = laneN. Column 15 = rowsums.
  if (laneN == 15) {
#pragma unroll
    for (int v = 0; v < 8; ++v) rowsum[w * 16 + v + 8 * hi] = d[v];
  }
  __syncthreads();

  // wave32 shuffle scan of the 16 rowsums (lanes 0..15 meaningful)
  float rs = rowsum[w * 16 + laneN];
  float inc = rs;
#pragma unroll
  for (int dl = 1; dl < 16; dl <<= 1) {
    float t = __shfl_up(inc, dl, 16);
    if (laneN >= dl) inc += t;
  }
  if (hi == 0) {
    rowoff[w * 16 + laneN] = inc - rs;       // exclusive row offset
    if (laneN == 15) waveoff[w] = inc;       // wave total (temp)
  }
  __syncthreads();

  if (tid == 0) {
    float run = 0.0f;
#pragma unroll
    for (int i = 0; i < 8; ++i) { float t = waveoff[i]; waveoff[i] = run; run += t; }
    partials[blockIdx.x] = run;              // block total
  }
  __syncthreads();

  float wo = waveoff[w];
#pragma unroll
  for (int v = 0; v < 8; ++v) {
    int i = v + 8 * hi;
    int li = wbase + i * 16 + laneN;
    int g = blkBase + li;
    if (g < NKEYS) {
      float excl = d[v] - occ[li] + rowoff[w * 16 + i] + wo;
      rank[g] = (int)excl;                   // block-local exclusive rank
    }
  }
}

// ---------------- K2b: single-block exclusive scan of partials -----------
__global__ __launch_bounds__(256) void k_scan2(float* __restrict__ partials,
                                               int* __restrict__ numUnique) {
  __shared__ float buf[256];
  __shared__ float buf2[256];
  int tid = threadIdx.x;
  float carry = 0.0f;
  for (int basei = 0; basei < NBLK2; basei += 256) {
    int i = basei + tid;
    float v = (i < NBLK2) ? partials[i] : 0.0f;
    float* src = buf;
    float* dst = buf2;
    src[tid] = v;
    __syncthreads();
#pragma unroll
    for (int d = 1; d < 256; d <<= 1) {
      float t = src[tid];
      if (tid >= d) t += src[tid - d];
      dst[tid] = t;
      __syncthreads();
      float* tmp = src; src = dst; dst = tmp;
    }
    float incl = src[tid];
    float total = src[255];
    if (i < NBLK2) partials[i] = carry + incl - v;   // exclusive + carry
    carry += total;
    __syncthreads();
  }
  if (tid == 0) *numUnique = (int)carry;
}

// ---------------- K2c: add global block offsets --------------------------
__global__ void k_addoff(int* __restrict__ rank, const float* __restrict__ partials) {
  int i = blockIdx.x * blockDim.x + threadIdx.x;
  int stride = gridDim.x * blockDim.x;
  for (; i < NKEYS; i += stride) rank[i] += (int)partials[i >> 11];  // 2048 = 2^11
}

// ---------------- K3: features + unq_inv ---------------------------------
// Async load staging for points; features assembled in LDS and drained with
// async LDS->global B128 stores (tile output region is 16B aligned:
// 512 pts * 36 B). S_ENDPGM's implicit wait-idle covers completion.
__global__ __launch_bounds__(256) void k_feat(const float* __restrict__ pts, int n,
                                              const uint32_t* __restrict__ cnt,
                                              const float* __restrict__ sx,
                                              const float* __restrict__ sy,
                                              const float* __restrict__ sz,
                                              const int* __restrict__ rank,
                                              float* __restrict__ feat,
                                              float* __restrict__ inv_out) {
  __shared__ float lds[TILE_PTS * 6];        // 12288 B in
  __shared__ float ofeat[TILE_PTS * 9];      // 18432 B out staging
  int tid = threadIdx.x;
  int base = blockIdx.x * TILE_PTS;
  int npts = n - base; if (npts > TILE_PTS) npts = TILE_PTS;
  if (npts <= 0) return;

  if (npts == TILE_PTS) {
    const char* g = (const char*)pts + (size_t)base * 24;
    char* l = (char*)lds;
#pragma unroll
    for (int k = 0; k < 3; ++k) {
      int off = tid * 16 + k * 4096;
      stage16(g + off, l + off);
    }
    wait_async0();
    __syncthreads();
  } else {
    for (int e = tid; e < npts * 6; e += 256) lds[e] = pts[(size_t)base * 6 + e];
    __syncthreads();
  }

  for (int p = tid; p < npts; p += 256) {
    const float* q = &lds[p * 6];
    int gidx = base + p;
    int cx, cy;
    int key = point_key(q, cx, cy);
    float cf = (float)cnt[key];
    if (cf < 1.0f) cf = 1.0f;
    float rc = 1.0f / cf;
    float mx = sx[key] * rc, my = sy[key] * rc, mz = sz[key] * rc;
    float* o = &ofeat[p * 9];
    o[0] = q[1]; o[1] = q[2]; o[2] = q[3]; o[3] = q[4];
    o[4] = q[1] - mx; o[5] = q[2] - my; o[6] = q[3] - mz;
    o[7] = q[1] - ((float)cx * 0.25f + 0.125f - 50.0f);
    o[8] = q[2] - ((float)cy * 0.25f + 0.125f - 50.0f);
    inv_out[gidx] = (float)rank[key];
  }
  __syncthreads();

  if (npts == TILE_PTS) {
    char* l = (char*)ofeat;
    char* gdst = (char*)(feat + (size_t)base * 9);
    for (int cch = tid; cch < (TILE_PTS * 36) / 16; cch += 256)   // 1152 chunks
      drain16(l + cch * 16, gdst + cch * 16);
  } else {
    for (int e = tid; e < npts * 9; e += 256) feat[(size_t)base * 9 + e] = ofeat[e];
  }
}

// ---------------- K4: decode occupied keys into unq rows -----------------
__global__ void k_unq(const uint32_t* __restrict__ cnt, const int* __restrict__ rank,
                      float* __restrict__ unq) {
  int g = blockIdx.x * blockDim.x + threadIdx.x;
  int stride = gridDim.x * blockDim.x;
  for (; g < NKEYS; g += stride) {
    if (cnt[g] != 0u) {
      int r = rank[g];
      int k = g;
      int t = k % NT; k /= NT;
      int y = k % GY; k /= GY;
      int x = k % GX; int b = k / GX;
      float* o = unq + (size_t)r * 4;
      o[0] = (float)b; o[1] = (float)t; o[2] = (float)y; o[3] = (float)x;
    }
  }
}

// ---------------- K5: -1 padding rows + GRID_SIZE ------------------------
__global__ void k_pad(float* __restrict__ unq, float* __restrict__ grid3,
                      const int* __restrict__ numUnique, int n) {
  int nu = *numUnique;
  int r = blockIdx.x * blockDim.x + threadIdx.x;
  int stride = gridDim.x * blockDim.x;
  for (; r < n; r += stride) {
    if (r >= nu) {
      float* o = unq + (size_t)r * 4;
      o[0] = -1.0f; o[1] = -1.0f; o[2] = -1.0f; o[3] = -1.0f;
    }
  }
  if (blockIdx.x == 0 && threadIdx.x == 0) {
    grid3[0] = 400.0f; grid3[1] = 400.0f; grid3[2] = 1.0f;
  }
}

// ---------------- host launcher ------------------------------------------
extern "C" void kernel_launch(void* const* d_in, const int* in_sizes, int n_in,
                              void* d_out, int out_size, void* d_ws, size_t ws_size,
                              hipStream_t stream) {
  const float* pts = (const float*)d_in[0];
  int n = in_sizes[0] / 6;

  char* ws = (char*)d_ws;
  uint32_t* cnt = (uint32_t*)ws;                         // NKEYS u32
  float* sx = (float*)(ws + (size_t)4 * NKEYS);          // NKEYS f32
  float* sy = sx + NKEYS;
  float* sz = sy + NKEYS;
  int* rank = (int*)(sz + NKEYS);                        // NKEYS i32
  float* partials = (float*)(rank + NKEYS);              // NBLK2 f32
  int* numU = (int*)(partials + NBLK2);                  // 1 i32
  // total ws use: ~64.1 MB

  float* feat  = (float*)d_out;                          // n*9
  float* unq   = feat + (size_t)n * 9;                   // n*4
  float* inv   = unq + (size_t)n * 4;                    // n
  float* grid3 = inv + n;                                // 3

  int ntiles = (n + TILE_PTS - 1) / TILE_PTS;
  int nacc = (ntiles + TPB - 1) / TPB;

  k_zero<<<4096, 256, 0, stream>>>((uint32_t*)ws, 4 * NKEYS);
  k_accum<<<nacc, 256, 0, stream>>>(pts, n, cnt, sx, sy, sz);
  k_scan1<<<NBLK2, 256, 0, stream>>>(cnt, rank, partials);
  k_scan2<<<1, 256, 0, stream>>>(partials, numU);
  k_addoff<<<2048, 256, 0, stream>>>(rank, partials);
  k_feat<<<ntiles, 256, 0, stream>>>(pts, n, cnt, sx, sy, sz, rank, feat, inv);
  k_unq<<<4096, 256, 0, stream>>>(cnt, rank, unq);
  k_pad<<<4096, 256, 0, stream>>>(unq, grid3, numU, n);
}
